// TemporalGraphNetwork_2319282340278
// MI455X (gfx1250) — matrix operand order repaired and checked by reading, not compiled, not run on hardware
//
#include <hip/hip_runtime.h>
#include <math.h>

// ---------------- problem constants (from reference) ----------------
#define NND   25000          // nodes
#define NE    200000         // edges
#define NP    25024          // NND rounded up to 32
#define MEMD  128
#define EVD   128
#define TEN   64
#define OUTD  128
#define NHEAD 4
#define HDIM  512            // HEADS*OUT
#define MSGD  464
#define MSGP  480            // MSGD rounded up to 32
// workspace requirement: ~620 MB (carved below)

typedef __attribute__((ext_vector_type(16))) __bf16 v16bf;
typedef __attribute__((ext_vector_type(8)))  float  v8f;

__device__ __forceinline__ unsigned short f2bf(float f) {
  unsigned u = __float_as_uint(f);
  u += 0x7FFFu + ((u >> 16) & 1u);          // round-to-nearest-even
  return (unsigned short)(u >> 16);
}
__device__ __forceinline__ float bf2f(unsigned short h) {
  return __uint_as_float(((unsigned)h) << 16);
}
// order-preserving float<->uint map for atomicMax on signed floats
__device__ __forceinline__ unsigned f2ord(float f) {
  unsigned u = __float_as_uint(f);
  return (u & 0x80000000u) ? ~u : (u | 0x80000000u);
}
__device__ __forceinline__ float ord2f(unsigned u) {
  unsigned b = (u & 0x80000000u) ? (u ^ 0x80000000u) : ~u;
  return __uint_as_float(b);
}
__device__ __forceinline__ float sigm(float x) { return 1.0f / (1.0f + expf(-x)); }

union FragU { uint4 u[2]; v16bf v; };

// ================= WMMA GEMM: C[N x M] = A[Npad x Kp](bf16) @ Bt[M x Kp]^T + bias =================
// A row-major bf16 (zero-padded rows/cols); Bt = weight transposed to [M x Kp] bf16.
// Per-lane fragment addressing follows the CDNA5 16-bit A(16x32)/B(32x16) VGPR layouts:
//   A: lane m=l&15; elems 0..7 -> K = 8*(l>>4)+e, elems 8..15 -> K = 16+8*(l>>4)+(e-8)
//   B: lane n=l&15; same K mapping as A
// Both are contiguous 16B chunks thanks to the transposed-B storage.
__global__ __launch_bounds__(128)
void gemm_bf16_wmma(const unsigned short* __restrict__ A,
                    const unsigned short* __restrict__ Bt,
                    const float* __restrict__ bias,
                    float* __restrict__ Cf,
                    unsigned short* __restrict__ Cb,
                    int Nrows, int Npad, int Kp, int M, int out_bf16)
{
  int wave = (blockIdx.x * blockDim.x + threadIdx.x) >> 5;
  int lane = threadIdx.x & 31;
  int tilesM = M >> 5;
  int tilesN = Npad >> 5;
  if (wave >= tilesM * tilesN) return;           // wave-uniform exit
  int tn = wave / tilesM, tm = wave % tilesM;
  int m0 = tn << 5, n0 = tm << 5;
  int hlf = lane >> 4;
  int l15 = lane & 15;

  const unsigned short* arow0 = A  + (size_t)(m0 + l15)      * Kp;
  const unsigned short* arow1 = A  + (size_t)(m0 + 16 + l15) * Kp;
  const unsigned short* brow0 = Bt + (size_t)(n0 + l15)      * Kp;
  const unsigned short* brow1 = Bt + (size_t)(n0 + 16 + l15) * Kp;

  v8f acc00 = {}, acc01 = {}, acc10 = {}, acc11 = {};
  for (int kk = 0; kk < Kp; kk += 32) {
    FragU a0, a1, b0, b1;
    a0.u[0] = *(const uint4*)(arow0 + kk + hlf * 8);
    a0.u[1] = *(const uint4*)(arow0 + kk + 16 + hlf * 8);
    a1.u[0] = *(const uint4*)(arow1 + kk + hlf * 8);
    a1.u[1] = *(const uint4*)(arow1 + kk + 16 + hlf * 8);
    b0.u[0] = *(const uint4*)(brow0 + kk + hlf * 8);
    b0.u[1] = *(const uint4*)(brow0 + kk + 16 + hlf * 8);
    b1.u[0] = *(const uint4*)(brow1 + kk + hlf * 8);
    b1.u[1] = *(const uint4*)(brow1 + kk + 16 + hlf * 8);
    acc00 = __builtin_amdgcn_wmma_f32_16x16x32_bf16(false, a0.v, false, b0.v, (short)0, acc00, false, false);
    acc01 = __builtin_amdgcn_wmma_f32_16x16x32_bf16(false, a0.v, false, b1.v, (short)0, acc01, false, false);
    acc10 = __builtin_amdgcn_wmma_f32_16x16x32_bf16(false, a1.v, false, b0.v, (short)0, acc10, false, false);
    acc11 = __builtin_amdgcn_wmma_f32_16x16x32_bf16(false, a1.v, false, b1.v, (short)0, acc11, false, false);
  }
  // D layout: elem r -> row = r + 8*(lane>>4), col = lane&15
  v8f accs[4] = { acc00, acc01, acc10, acc11 };
  #pragma unroll
  for (int i = 0; i < 2; i++) {
    #pragma unroll
    for (int j = 0; j < 2; j++) {
      v8f a = accs[i * 2 + j];
      int col = n0 + 16 * j + l15;
      float bv = bias ? bias[col] : 0.0f;
      #pragma unroll
      for (int r = 0; r < 8; r++) {
        int row = m0 + 16 * i + r + 8 * hlf;
        if (row < Nrows) {
          float val = a[r] + bv;
          if (out_bf16) Cb[(size_t)row * M + col] = f2bf(val);
          else          Cf[(size_t)row * M + col] = val;
        }
      }
    }
  }
}

// ============ weight transpose + bf16 convert: W[K x M] -> Wt[M x Kp] (zero pad k>=K) ============
__global__ void wtrans_bf16(const float* __restrict__ W, unsigned short* __restrict__ Wt,
                            int K, int M, int Kp)
{
  int idx = blockIdx.x * blockDim.x + threadIdx.x;
  if (idx >= M * Kp) return;
  int m = idx / Kp, k = idx - m * Kp;
  float v = (k < K) ? W[(size_t)k * M + m] : 0.0f;
  Wt[idx] = f2bf(v);
}

// ==== concat up to 3 f32 row-major sources into padded bf16 activation matrix [Npad x Kp] ====
__global__ void concat_bf16(const float* __restrict__ S0, int K0,
                            const float* __restrict__ S1, int K1,
                            const float* __restrict__ S2, int K2,
                            unsigned short* __restrict__ D,
                            int Nrows, int Npad, int Kp)
{
  int idx = blockIdx.x * blockDim.x + threadIdx.x;
  if (idx >= Npad * Kp) return;
  int n = idx / Kp, k = idx - n * Kp;
  float v = 0.0f;
  if (n < Nrows) {
    if (k < K0)                v = S0[(size_t)n * K0 + k];
    else if (k < K0 + K1)      v = S1[(size_t)n * K1 + (k - K0)];
    else if (k < K0 + K1 + K2) v = S2[(size_t)n * K2 + (k - K0 - K1)];
  }
  D[idx] = f2bf(v);
}

// ======================== GRU gate fusion: um = (1-z)*n + z*h ========================
__global__ void gru_update(const float* __restrict__ gi, const float* __restrict__ gh,
                           const float* __restrict__ h, float* __restrict__ um, int Nn)
{
  int idx = blockIdx.x * blockDim.x + threadIdx.x;
  if (idx >= Nn * MEMD) return;
  int n = idx >> 7, j = idx & 127;
  const float* gin = gi + (size_t)n * 384;
  const float* ghn = gh + (size_t)n * 384;
  float r  = sigm(gin[j]       + ghn[j]);
  float z  = sigm(gin[128 + j] + ghn[128 + j]);
  float nn = tanhf(gin[256 + j] + r * ghn[256 + j]);
  um[idx] = (1.0f - z) * nn + z * h[(size_t)n * MEMD + j];
}

// ===== edge features: ef = [cos(rel_t*w+b), edge_attr] as bf16 [E x 192] =====
__global__ void build_ef(const int* __restrict__ src, const float* __restrict__ eattr,
                         const float* __restrict__ elu, const int* __restrict__ nbatch,
                         const float* __restrict__ ets, const float* __restrict__ tw,
                         const float* __restrict__ tb, unsigned short* __restrict__ ef, int E)
{
  int idx = blockIdx.x * blockDim.x + threadIdx.x;
  if (idx >= E * 192) return;
  int e = idx / 192, j = idx - e * 192;
  float v;
  if (j < TEN) {
    float rt = ets[nbatch[src[e]]] - elu[e];
    v = cosf(rt * tw[j] + tb[j]);
  } else {
    v = eattr[(size_t)e * EVD + (j - TEN)];
  }
  ef[idx] = f2bf(v);
}

__global__ void init_attn(unsigned int* __restrict__ lmax, float* __restrict__ denom, int n)
{
  int idx = blockIdx.x * blockDim.x + threadIdx.x;
  if (idx >= n) return;
  lmax[idx]  = 0x007FFFFFu;   // f2ord(-inf)
  denom[idx] = 0.0f;
}

// ===== pass A: per-edge logits + segment max (one wave per edge, 16 dims/lane) =====
__global__ __launch_bounds__(256)
void attn_logits(const int* __restrict__ src, const int* __restrict__ dst,
                 const float* __restrict__ q, const float* __restrict__ k,
                 const unsigned short* __restrict__ ebf,
                 float* __restrict__ logits, unsigned int* __restrict__ lmax, int E)
{
  int e = (blockIdx.x * blockDim.x + threadIdx.x) >> 5;
  int lane = threadIdx.x & 31;
  if (e >= E) return;
  int s = src[e], d = dst[e];
  alignas(16) float qa[16], ka[16];
  alignas(16) unsigned short ea[16];
  const float4* qp = (const float4*)(q + (size_t)d * HDIM + lane * 16);
  const float4* kp = (const float4*)(k + (size_t)s * HDIM + lane * 16);
  const uint4*  ep = (const uint4*)(ebf + (size_t)e * HDIM + lane * 16);
  #pragma unroll
  for (int i = 0; i < 4; i++) { *(float4*)(qa + 4 * i) = qp[i]; *(float4*)(ka + 4 * i) = kp[i]; }
  *(uint4*)(ea) = ep[0]; *(uint4*)(ea + 8) = ep[1];
  float sum = 0.0f;
  #pragma unroll
  for (int i = 0; i < 16; i++) sum += qa[i] * (ka[i] + bf2f(ea[i]));
  sum += __shfl_xor(sum, 1, 32);
  sum += __shfl_xor(sum, 2, 32);
  sum += __shfl_xor(sum, 4, 32);                 // 8-lane group == one head (128 dims)
  if ((lane & 7) == 0) {
    int hh = lane >> 3;
    float logit = sum * 0.08838834764831845f;    // 1/sqrt(OUT=128)
    logits[(size_t)e * NHEAD + hh] = logit;
    atomicMax(&lmax[(size_t)d * NHEAD + hh], f2ord(logit));
  }
}

// ===== pass B: segment softmax denominator =====
__global__ void attn_denom(const int* __restrict__ dst, const float* __restrict__ logits,
                           const unsigned int* __restrict__ lmax, float* __restrict__ denom, int E)
{
  int idx = blockIdx.x * blockDim.x + threadIdx.x;
  if (idx >= E * NHEAD) return;
  int e = idx >> 2, hh = idx & 3;
  int d = dst[e];
  float a = expf(logits[idx] - ord2f(lmax[(size_t)d * NHEAD + hh]));
  atomicAdd(&denom[(size_t)d * NHEAD + hh], a);
}

// ===== pass C: msg = (v[src]+e)*alpha scatter-added onto skip-initialized output =====
__global__ __launch_bounds__(256)
void attn_aggregate(const int* __restrict__ src, const int* __restrict__ dst,
                    const float* __restrict__ v, const unsigned short* __restrict__ ebf,
                    const float* __restrict__ logits, const unsigned int* __restrict__ lmax,
                    const float* __restrict__ denom, float* __restrict__ emb, int E)
{
  int e = (blockIdx.x * blockDim.x + threadIdx.x) >> 5;
  int lane = threadIdx.x & 31;
  if (e >= E) return;
  int s = src[e], d = dst[e];
  int hh = lane >> 3;
  float lm  = ord2f(lmax[(size_t)d * NHEAD + hh]);
  float den = denom[(size_t)d * NHEAD + hh];
  float alpha = expf(logits[(size_t)e * NHEAD + hh] - lm) / (den + 1e-16f);
  alignas(16) float va[16];
  alignas(16) unsigned short ea[16];
  const float4* vp = (const float4*)(v + (size_t)s * HDIM + lane * 16);
  const uint4*  ep = (const uint4*)(ebf + (size_t)e * HDIM + lane * 16);
  #pragma unroll
  for (int i = 0; i < 4; i++) *(float4*)(va + 4 * i) = vp[i];
  *(uint4*)(ea) = ep[0]; *(uint4*)(ea + 8) = ep[1];
  float* out = emb + (size_t)d * HDIM + lane * 16;
  #pragma unroll
  for (int i = 0; i < 16; i++) atomicAdd(out + i, (va[i] + bf2f(ea[i])) * alpha);
}

// ============================== host orchestration ==============================
extern "C" void kernel_launch(void* const* d_in, const int* in_sizes, int n_in,
                              void* d_out, int out_size, void* d_ws, size_t ws_size,
                              hipStream_t stream)
{
  (void)in_sizes; (void)n_in; (void)out_size; (void)ws_size;
  // input order: x, edge_index, edge_attr, edge_last_update, node_batch, event_timestamps,
  // memory, messages, gru{w_ih,w_hh,b_ih,b_hh}, time_w, time_b,
  // 3x gnn_blocks{wq,bq,wk,bk,wv,bv,we,ws,bs}, out_w, out_b
  const float* x    = (const float*)d_in[0];
  const int*   ei   = (const int*)  d_in[1];
  const float* eat  = (const float*)d_in[2];
  const float* elu  = (const float*)d_in[3];
  const int*   nb   = (const int*)  d_in[4];
  const float* ets  = (const float*)d_in[5];
  const float* mem  = (const float*)d_in[6];
  const float* msgs = (const float*)d_in[7];
  const float* w_ih = (const float*)d_in[8];
  const float* w_hh = (const float*)d_in[9];
  const float* b_ih = (const float*)d_in[10];
  const float* b_hh = (const float*)d_in[11];
  const float* tw   = (const float*)d_in[12];
  const float* tb   = (const float*)d_in[13];
  const float* outw = (const float*)d_in[41];
  const float* outb = (const float*)d_in[42];
  const int* src = ei;
  const int* dst = ei + NE;

  // ---- workspace carve (bump allocator, 256B aligned) ----
  char* base = (char*)d_ws;
  size_t off = 0;
  auto carve = [&](size_t bytes) -> void* {
    void* r = base + off;
    off += (bytes + 255) & ~(size_t)255;
    return r;
  };
  float*          umem   = (float*)carve((size_t)NND * MEMD * 4);
  unsigned short* efb    = (unsigned short*)carve((size_t)NE * 192 * 2);
  unsigned short* ebf    = (unsigned short*)carve((size_t)NE * HDIM * 2);
  float*          qb     = (float*)carve((size_t)NND * HDIM * 4);
  float*          kb     = (float*)carve((size_t)NND * HDIM * 4);
  float*          vb     = (float*)carve((size_t)NND * HDIM * 4);
  float*          embA   = (float*)carve((size_t)NND * HDIM * 4);
  float*          embB   = (float*)carve((size_t)NND * HDIM * 4);
  unsigned short* hinb   = (unsigned short*)carve((size_t)NP * 768 * 2); // also msg_bf / final emb_bf
  unsigned short* memb   = (unsigned short*)carve((size_t)NP * MEMD * 2);
  unsigned short* wt     = (unsigned short*)carve((size_t)512 * 768 * 2);
  float*          logits = (float*)carve((size_t)NE * NHEAD * 4);
  unsigned int*   lmax   = (unsigned int*)carve((size_t)NND * NHEAD * 4);
  float*          denom  = (float*)carve((size_t)NND * NHEAD * 4);
  float* gi = qb;  // gemm outputs [NND x 384] fit inside q/k buffers; dead before q/k written
  float* gh = kb;

  auto launch_wt = [&](const float* W, int K, int M, int Kp) {
    int total = M * Kp;
    wtrans_bf16<<<(total + 255) / 256, 256, 0, stream>>>(W, wt, K, M, Kp);
  };
  auto launch_cat = [&](const float* S0, int K0, const float* S1, int K1,
                        const float* S2, int K2, unsigned short* D, int Kp) {
    int total = NP * Kp;
    concat_bf16<<<(total + 255) / 256, 256, 0, stream>>>(S0, K0, S1, K1, S2, K2, D, NND, NP, Kp);
  };
  auto launch_gemm = [&](const unsigned short* A, const float* bias, float* Cf,
                         unsigned short* Cb, int Nrows, int Npad, int Kp, int M) {
    int waves = (Npad / 32) * (M / 32);
    gemm_bf16_wmma<<<(waves + 3) / 4, 128, 0, stream>>>(A, wt, bias, Cf, Cb,
                                                        Nrows, Npad, Kp, M, Cb != nullptr);
  };

  // ---- GRU memory update ----
  launch_cat(msgs, MSGD, nullptr, 0, nullptr, 0, hinb, MSGP);
  launch_wt(w_ih, MSGD, 384, MSGP);
  launch_gemm(hinb, b_ih, gi, nullptr, NND, NP, MSGP, 384);
  launch_cat(mem, MEMD, nullptr, 0, nullptr, 0, memb, MEMD);
  launch_wt(w_hh, MEMD, 384, MEMD);
  launch_gemm(memb, b_hh, gh, nullptr, NND, NP, MEMD, 384);
  gru_update<<<(NND * MEMD + 255) / 256, 256, 0, stream>>>(gi, gh, mem, umem, NND);

  // ---- edge features (time encoding + attr), bf16 ----
  build_ef<<<(NE * 192 + 255) / 256, 256, 0, stream>>>(src, eat, elu, nb, ets, tw, tb, efb, NE);

  // ---- 3 TransformerConv blocks ----
  float* prev = nullptr;
  float* bufs[2] = { embA, embB };
  for (int i = 0; i < 3; i++) {
    int bi = 14 + 9 * i;
    const float* wq  = (const float*)d_in[bi + 0];
    const float* bq  = (const float*)d_in[bi + 1];
    const float* wk  = (const float*)d_in[bi + 2];
    const float* bk  = (const float*)d_in[bi + 3];
    const float* wv  = (const float*)d_in[bi + 4];
    const float* bv  = (const float*)d_in[bi + 5];
    const float* we  = (const float*)d_in[bi + 6];
    const float* wsk = (const float*)d_in[bi + 7];
    const float* bsk = (const float*)d_in[bi + 8];
    int ind = (i == 0) ? 256 : 768;
    float* outbuf = bufs[i & 1];

    if (i == 0) launch_cat(x, EVD, umem, MEMD, nullptr, 0, hinb, 256);
    else        launch_cat(prev, HDIM, x, EVD, umem, MEMD, hinb, 768);

    launch_wt(wq, ind, HDIM, ind);  launch_gemm(hinb, bq, qb, nullptr, NND, NP, ind, HDIM);
    launch_wt(wk, ind, HDIM, ind);  launch_gemm(hinb, bk, kb, nullptr, NND, NP, ind, HDIM);
    launch_wt(wv, ind, HDIM, ind);  launch_gemm(hinb, bv, vb, nullptr, NND, NP, ind, HDIM);
    launch_wt(wsk, ind, HDIM, ind); launch_gemm(hinb, bsk, outbuf, nullptr, NND, NP, ind, HDIM); // skip init
    launch_wt(we, 192, HDIM, 192);  launch_gemm(efb, nullptr, nullptr, ebf, NE, NE, 192, HDIM);  // e, bf16

    init_attn<<<(NND * NHEAD + 255) / 256, 256, 0, stream>>>(lmax, denom, NND * NHEAD);
    attn_logits<<<(NE + 7) / 8, 256, 0, stream>>>(src, dst, qb, kb, ebf, logits, lmax, NE);
    attn_denom<<<(NE * NHEAD + 255) / 256, 256, 0, stream>>>(dst, logits, lmax, denom, NE);
    attn_aggregate<<<(NE + 7) / 8, 256, 0, stream>>>(src, dst, vb, ebf, logits, lmax, denom, outbuf, NE);
    prev = outbuf;
  }

  // ---- output projection ----
  launch_cat(prev, HDIM, nullptr, 0, nullptr, 0, hinb, HDIM);
  launch_wt(outw, HDIM, OUTD, HDIM);
  launch_gemm(hinb, outb, (float*)d_out, nullptr, NND, NP, HDIM, OUTD);
}
